// GraphClassificationModel_13692355740146
// MI455X (gfx1250) — compile-verified, matrix operand order checked by reading, not run on hardware
//
#include <hip/hip_runtime.h>
#include <math.h>

// ---------------------------------------------------------------------------
// GCN graph-classification forward for MI455X (gfx1250, wave32, WMMA).
// GEMMs: v_wmma_f32_16x16x32_bf16, 16x64 tile per wave.
// Scatter-mean: one wave per edge, fp32 global atomics (bandwidth bound).
// ---------------------------------------------------------------------------

#define F 256                 // feature width of every GCN layer

typedef __attribute__((ext_vector_type(16))) __bf16        v16bf;
typedef __attribute__((ext_vector_type(8)))  float         v8f;
typedef __attribute__((ext_vector_type(8)))  unsigned short u16x8;

union FragBF {
    v16bf v;
    u16x8 h[2];
};

__device__ __forceinline__ unsigned short f2bf(float f) {
    unsigned int u = __float_as_uint(f);
    unsigned int r = u + 0x7FFFu + ((u >> 16) & 1u);   // round-to-nearest-even
    return (unsigned short)(r >> 16);
}
__device__ __forceinline__ float bf2f(unsigned short h) {
    return __uint_as_float(((unsigned int)h) << 16);
}

// ------------------------------- utilities ---------------------------------

__global__ __launch_bounds__(256) void zero_f32(float* __restrict__ p, size_t n) {
    size_t i = (size_t)blockIdx.x * blockDim.x + threadIdx.x;
    if (i < n) p[i] = 0.0f;
}

__global__ __launch_bounds__(256) void count_deg(const int* __restrict__ colI,
                                                 float* __restrict__ cnt, int E) {
    int e = blockIdx.x * blockDim.x + threadIdx.x;
    if (e < E) atomicAdd(&cnt[colI[e]], 1.0f);
}

// fp32 [N*F] -> bf16 [N*F]   (4 floats / thread)
__global__ __launch_bounds__(256) void convert_bf16(const float4* __restrict__ src,
                                                    uint2* __restrict__ dst, size_t n4) {
    size_t i = (size_t)blockIdx.x * blockDim.x + threadIdx.x;
    if (i >= n4) return;
    float4 v = src[i];
    uint2 o;
    o.x = (unsigned)f2bf(v.x) | ((unsigned)f2bf(v.y) << 16);
    o.y = (unsigned)f2bf(v.z) | ((unsigned)f2bf(v.w) << 16);
    dst[i] = o;
}

// -------------------------- weight repack (B frag) -------------------------
// wpack layout: [(ct*8 + ks)*32 + lane] * 16 contiguous bf16.
// Element j of a lane's fragment (column n = ct*16 + lane%16, halfsel = lane/16):
//   j <  8 : K = ks*32 + 8*halfsel + j
//   j >= 8 : K = ks*32 + 8*halfsel + 8 + j      (i.e. +16 over first chunk)
__global__ __launch_bounds__(256) void pack_w(const float* __restrict__ W,
                                              unsigned short* __restrict__ wpack) {
    int idx  = blockIdx.x * 256 + threadIdx.x;    // 0 .. 65535
    int j    = idx & 15;
    int lane = (idx >> 4) & 31;
    int ks   = (idx >> 9) & 7;
    int ct   = idx >> 12;
    int half = lane >> 4;
    int n    = ct * 16 + (lane & 15);
    int k    = ks * 32 + 8 * half + ((j < 8) ? j : (8 + j));
    wpack[idx] = f2bf(W[k * F + n]);
}

// --------------------------------- GEMM ------------------------------------
// lin[N x 256] = bf16(h)[N x 256] @ W[256 x 256] + b   (fp32 out)
// Block = 256 threads = 8 waves; each wave: 16 rows x 64 cols.
__global__ __launch_bounds__(256) void gemm_bf16(const unsigned short* __restrict__ hbf,
                                                 const unsigned short* __restrict__ wpack,
                                                 const float* __restrict__ bias,
                                                 float* __restrict__ lin, int N) {
    int tid   = threadIdx.x;
    int wave  = tid >> 5;                        // 0..7
    int lane  = tid & 31;
    int rowTile = blockIdx.x * 2 + (wave >> 2);  // 2 row-tiles per block
    int cg      = wave & 3;                      // column group: 64 cols
    int rowBase = rowTile * 16;
    if (rowBase >= N) return;                    // wave-uniform guard

    int half = lane >> 4;
    int m    = rowBase + (lane & 15);
    int mc   = (m < N) ? m : (N - 1);            // clamp loads, keep EXEC full
    const unsigned short* arow = hbf + (size_t)mc * F;

    v8f acc[4] = {};
    #pragma unroll
    for (int ks = 0; ks < 8; ++ks) {
        int k0 = ks * 32;
        FragBF a;
        a.h[0] = *(const u16x8*)(arow + k0 + 8 * half);
        a.h[1] = *(const u16x8*)(arow + k0 + 8 * half + 16);
        #pragma unroll
        for (int c = 0; c < 4; ++c) {
            int ct = cg * 4 + c;
            FragBF b;
            const unsigned short* bp = wpack + (((ct * 8 + ks) * 32 + lane) << 4);
            b.h[0] = *(const u16x8*)(bp);
            b.h[1] = *(const u16x8*)(bp + 8);
            acc[c] = __builtin_amdgcn_wmma_f32_16x16x32_bf16(
                false, a.v, false, b.v, (short)0, acc[c], false, false);
        }
    }

    // C/D layout: VGPR r holds M = r + 8*half, N = lane%15.. (lane&15)
    int ncol = lane & 15;
    #pragma unroll
    for (int c = 0; c < 4; ++c) {
        int col = (cg * 4 + c) * 16 + ncol;
        float bv = bias[col];
        #pragma unroll
        for (int r = 0; r < 8; ++r) {
            int mm = rowBase + r + 8 * half;
            if (mm < N) lin[(size_t)mm * F + col] = acc[c][r] + bv;
        }
    }
}

// ----------------------------- edge scatter --------------------------------
// One wave per edge: summed[col[e]][:] += lin[row[e]][:]
__global__ __launch_bounds__(256) void scatter_edges(const float* __restrict__ lin,
                                                     const int* __restrict__ rowI,
                                                     const int* __restrict__ colI,
                                                     float* __restrict__ summed, int E) {
    int wave = blockIdx.x * 8 + (threadIdx.x >> 5);
    int lane = threadIdx.x & 31;
    if (wave >= E) return;
    int r = rowI[wave];
    int c = colI[wave];
    const float4* src = (const float4*)(lin + (size_t)r * F);
    float* dst = summed + (size_t)c * F + lane * 8;
    #pragma unroll
    for (int i = 0; i < 2; ++i) {
        float4 v = src[lane * 2 + i];
        float* d = dst + i * 4;
        atomicAdd(d + 0, v.x);
        atomicAdd(d + 1, v.y);
        atomicAdd(d + 2, v.z);
        atomicAdd(d + 3, v.w);
    }
}

// ------------------------ mean / fallback / relu / cvt ---------------------
__global__ __launch_bounds__(256) void combine_relu(const float4* __restrict__ lin4,
                                                    const float4* __restrict__ sum4,
                                                    const float* __restrict__ cnt,
                                                    uint2* __restrict__ hbf, size_t n4) {
    size_t i = (size_t)blockIdx.x * blockDim.x + threadIdx.x;
    if (i >= n4) return;
    int node = (int)(i >> 6);                    // 64 float4 groups per node
    float c  = cnt[node];
    float inv = 1.0f / fmaxf(c, 1.0f);
    float4 l = lin4[i];
    float4 s = sum4[i];
    float4 o;
    o.x = fmaxf((c > 0.0f) ? s.x * inv : l.x, 0.0f);
    o.y = fmaxf((c > 0.0f) ? s.y * inv : l.y, 0.0f);
    o.z = fmaxf((c > 0.0f) ? s.z * inv : l.z, 0.0f);
    o.w = fmaxf((c > 0.0f) ? s.w * inv : l.w, 0.0f);
    uint2 p;
    p.x = (unsigned)f2bf(o.x) | ((unsigned)f2bf(o.y) << 16);
    p.y = (unsigned)f2bf(o.z) | ((unsigned)f2bf(o.w) << 16);
    hbf[i] = p;
}

// ------------------------------ global pool --------------------------------
// Each block: thread t owns feature t, all threads sweep the same row range
// (fully coalesced row reads), one atomic per (block, feature).
__global__ __launch_bounds__(256) void pool_sum(const unsigned short* __restrict__ hbf,
                                                float* __restrict__ gsum, int N, int rpb) {
    int f  = threadIdx.x;
    int i0 = blockIdx.x * rpb;
    int i1 = i0 + rpb;
    if (i1 > N) i1 = N;
    float acc = 0.0f;
    for (int i = i0; i < i1; ++i) acc += bf2f(hbf[(size_t)i * F + f]);
    if (acc != 0.0f || i0 < i1) atomicAdd(&gsum[f], acc);
}

// ------------------------------ classifier ---------------------------------
__global__ __launch_bounds__(128) void classify(const float* __restrict__ gsum,
                                                const float* __restrict__ Wc1,
                                                const float* __restrict__ bc1,
                                                const float* __restrict__ Wc2,
                                                const float* __restrict__ bc2,
                                                float* __restrict__ out, float invN) {
    __shared__ float g[256];
    __shared__ float z[128];
    __shared__ float logits[10];
    int t = threadIdx.x;
    g[t]       = gsum[t] * invN;
    g[t + 128] = gsum[t + 128] * invN;
    __syncthreads();
    float a = bc1[t];
    #pragma unroll 4
    for (int k = 0; k < 256; ++k) a += g[k] * Wc1[k * 128 + t];
    z[t] = fmaxf(a, 0.0f);
    __syncthreads();
    if (t < 10) {
        float l = bc2[t];
        for (int j = 0; j < 128; ++j) l += z[j] * Wc2[j * 10 + t];
        logits[t] = l;
    }
    __syncthreads();
    if (t == 0) {
        float mx = logits[0];
        for (int c = 1; c < 10; ++c) mx = fmaxf(mx, logits[c]);
        float se = 0.0f;
        for (int c = 0; c < 10; ++c) se += __expf(logits[c] - mx);
        float lse = mx + __logf(se);
        for (int c = 0; c < 10; ++c) out[c] = logits[c] - lse;
    }
}

// ------------------------------- launcher ----------------------------------

extern "C" void kernel_launch(void* const* d_in, const int* in_sizes, int n_in,
                              void* d_out, int out_size, void* d_ws, size_t ws_size,
                              hipStream_t stream) {
    const float* x   = (const float*)d_in[0];
    const int*   ei  = (const int*)d_in[1];     // harness contract: integer -> int*
    const float* Ws[3] = {(const float*)d_in[2], (const float*)d_in[4], (const float*)d_in[6]};
    const float* bs[3] = {(const float*)d_in[3], (const float*)d_in[5], (const float*)d_in[7]};
    const float* Wc1 = (const float*)d_in[8];
    const float* bc1 = (const float*)d_in[9];
    const float* Wc2 = (const float*)d_in[10];
    const float* bc2 = (const float*)d_in[11];
    float* out = (float*)d_out;

    const int N = in_sizes[0] / F;
    const int E = in_sizes[1] / 2;
    const int* rowI = ei;
    const int* colI = ei + E;

    // workspace carve-out
    char*  ws  = (char*)d_ws;
    size_t off = 0;
    auto carve = [&](size_t bytes) { size_t p = off; off += (bytes + 255) & ~(size_t)255; return p; };
    float*          lin    = (float*)(ws + carve((size_t)N * F * 4));
    float*          summed = (float*)(ws + carve((size_t)N * F * 4));
    unsigned short* hbf    = (unsigned short*)(ws + carve((size_t)N * F * 2));
    float*          cnt    = (float*)(ws + carve((size_t)N * 4));
    unsigned short* wpack  = (unsigned short*)(ws + carve((size_t)65536 * 2));
    float*          gsum   = (float*)(ws + carve(256 * 4));
    (void)ws_size; (void)n_in; (void)out_size;

    const size_t nF  = (size_t)N * F;            // 25.6M
    const size_t n4  = nF / 4;                   // float4 groups
    dim3 blk(256);

    // degree counts (identical for all layers) + zero accumulators
    zero_f32<<<(unsigned)((N + 255) / 256), blk, 0, stream>>>(cnt, (size_t)N);
    zero_f32<<<1, blk, 0, stream>>>(gsum, 256);
    count_deg<<<(unsigned)((E + 255) / 256), blk, 0, stream>>>(colI, cnt, E);

    // x -> bf16 activations
    convert_bf16<<<(unsigned)((n4 + 255) / 256), blk, 0, stream>>>(
        (const float4*)x, (uint2*)hbf, n4);

    for (int L = 0; L < 3; ++L) {
        pack_w<<<256, blk, 0, stream>>>(Ws[L], wpack);
        gemm_bf16<<<(unsigned)((N + 31) / 32), blk, 0, stream>>>(hbf, wpack, bs[L], lin, N);
        zero_f32<<<(unsigned)((nF + 255) / 256), blk, 0, stream>>>(summed, nF);
        scatter_edges<<<(unsigned)((E + 7) / 8), blk, 0, stream>>>(lin, rowI, colI, summed, E);
        combine_relu<<<(unsigned)((n4 + 255) / 256), blk, 0, stream>>>(
            (const float4*)lin, (const float4*)summed, cnt, (uint2*)hbf, n4);
    }

    int rpb = (N + 399) / 400;
    pool_sum<<<400, blk, 0, stream>>>(hbf, gsum, N, rpb);
    classify<<<1, 128, 0, stream>>>(gsum, Wc1, bc1, Wc2, bc2, out, 1.0f / (float)N);
}
